// indicator_layer_19318762897747
// MI455X (gfx1250) — matrix operand order repaired and checked by reading, not compiled, not run on hardware
//
#include <hip/hip_runtime.h>
#include <hip/hip_bf16.h>

typedef __attribute__((ext_vector_type(16))) _Float16 v16h;
typedef __attribute__((ext_vector_type(8)))  float    v8f;
typedef __attribute__((ext_vector_type(4)))  float    v4f;

#define NB    2
#define NPTS  8192
#define DIM   64
#define KNN_K 16

// ---- workspace layout (bytes) ----
#define WS_F    0u
#define WS_IDX  (4u * 1024 * 1024)
#define WS_WPP  (5u * 1024 * 1024)
#define WS_W2P  (5u * 1024 * 1024 + 8192u)

#define WMMA_F16(A, Bv, C) \
  __builtin_amdgcn_wmma_f32_16x16x32_f16(false, (A), false, (Bv), (short)0, (C), false, false)

// ----------------------------------------------------------------------------
// Pack a 64x64 f32 weight matrix into the wave32 f16 B-operand register layout
// for v_wmma_f32_16x16x32_f16 (lane<16: K=0..7,16..23 of chunk for col n=lane;
// lanes 16..31: K=8..15,24..31).
// ----------------------------------------------------------------------------
__global__ void __launch_bounds__(256)
pack_b16(const float* __restrict__ W, _Float16* __restrict__ dst) {
  int e = blockIdx.x * blockDim.x + threadIdx.x;
  if (e >= 4096) return;
  int v    = e & 15;
  int lane = (e >> 4) & 31;
  int qt   = e >> 9;                 // q*4 + t
  int q = qt >> 2, t = qt & 3;
  int hi = lane >> 4, n = lane & 15;
  int K = q * 32 + (v < 8 ? hi * 8 + v : 16 + hi * 8 + (v - 8));
  int c = t * 16 + n;
  dst[e] = (_Float16)W[K * DIM + c];
}

// ----------------------------------------------------------------------------
// Kernel 1: f = feature1 @ Wp + bp  (16384x64, K=64). One wave = 16-row stripe,
// 4 col tiles x 2 K-chunks = 8 WMMAs. A built directly in ISA f16 A layout.
// ----------------------------------------------------------------------------
__global__ void __launch_bounds__(256)
project_wmma(const float* __restrict__ feat, const _Float16* __restrict__ WpP,
             const float* __restrict__ bp, float* __restrict__ fout) {
  const int lane = threadIdx.x & 31;
  const int hi = lane >> 4, n = lane & 15;
  const int wave = (blockIdx.x * blockDim.x + threadIdx.x) >> 5;
  const int rowBase = wave * 16;
  const float* frow = feat + (size_t)(rowBase + n) * DIM;

  v16h a[2];
#pragma unroll
  for (int c = 0; c < 2; ++c) {
    v4f r0 = *(const v4f*)(frow + c * 32 + hi * 8);
    v4f r1 = *(const v4f*)(frow + c * 32 + hi * 8 + 4);
    v4f r2 = *(const v4f*)(frow + c * 32 + 16 + hi * 8);
    v4f r3 = *(const v4f*)(frow + c * 32 + 16 + hi * 8 + 4);
#pragma unroll
    for (int e = 0; e < 4; ++e) {
      a[c][e]      = (_Float16)r0[e];
      a[c][4 + e]  = (_Float16)r1[e];
      a[c][8 + e]  = (_Float16)r2[e];
      a[c][12 + e] = (_Float16)r3[e];
    }
  }

#pragma unroll
  for (int t = 0; t < 4; ++t) {
    v16h b0 = *(const v16h*)(WpP + ((size_t)(0 * 4 + t) * 32 + lane) * 16);
    v16h b1 = *(const v16h*)(WpP + ((size_t)(1 * 4 + t) * 32 + lane) * 16);
    v8f acc = {};
    acc = WMMA_F16(a[0], b0, acc);
    acc = WMMA_F16(a[1], b1, acc);
    const float bias = bp[t * 16 + n];
#pragma unroll
    for (int r = 0; r < 8; ++r)
      fout[(size_t)(rowBase + r + hi * 8) * DIM + t * 16 + n] = acc[r] + bias;
  }
}

// ----------------------------------------------------------------------------
// Kernel 2: brute-force 16-NN, one lane per query. xyz1 is streamed through
// LDS with double-buffered GLOBAL_LOAD_ASYNC_TO_LDS_B128 (ASYNCcnt-tracked,
// no VGPR round-trip): chunk c+1 fills while chunk c's distance/insert loop
// runs, so the 96 KB/block slab re-read is fully hidden behind VALU work.
// ----------------------------------------------------------------------------
#define CHUNK       2048
#define CHUNK_BYTES (CHUNK * 3 * 4)        // 24576 B
#define NCHUNK      (NPTS / CHUNK)         // 4
#define ASYNC_PASSES 6                     // 256 thr * 16 B * 6 = 24576 B

__device__ __forceinline__ void async_stage_chunk(uint64_t gbase, uint32_t lbase, int tid) {
  uint64_t g = gbase + (uint64_t)tid * 16u;
  uint32_t l = lbase + (uint32_t)tid * 16u;
#pragma unroll
  for (int p = 0; p < ASYNC_PASSES; ++p) {
    asm volatile("global_load_async_to_lds_b128 %0, %1, off"
                 :: "v"(l + (uint32_t)(p * 4096)), "v"(g + (uint64_t)(p * 4096))
                 : "memory");
  }
}

__global__ void __launch_bounds__(256)
knn_bruteforce(const float* __restrict__ xyz1, const float* __restrict__ xyz2,
               int* __restrict__ idx_out, float* __restrict__ weight_out) {
  __shared__ __align__(16) float sx[2][CHUNK * 3];
  const int tid = threadIdx.x;
  const int m = blockIdx.x * 256 + tid;
  const int b = (blockIdx.x * 256) >> 13;            // block-uniform batch
  const uint64_t gbase = (uint64_t)(uintptr_t)(xyz1 + (size_t)b * NPTS * 3);
  // flat LDS addresses carry the LDS byte offset in addr[31:0]
  const uint32_t lds0 = (uint32_t)(uintptr_t)&sx[0][0];
  const uint32_t lds1 = (uint32_t)(uintptr_t)&sx[1][0];

  const float qx = xyz2[(size_t)m * 3 + 0];
  const float qy = xyz2[(size_t)m * 3 + 1];
  const float qz = xyz2[(size_t)m * 3 + 2];

  float bd[KNN_K];
  int   bi[KNN_K];
#pragma unroll
  for (int k = 0; k < KNN_K; ++k) { bd[k] = 3.4e38f; bi[k] = 0; }

  // prologue: fill buffer 0 with chunk 0
  async_stage_chunk(gbase, lds0, tid);
  asm volatile("s_wait_asynccnt 0" ::: "memory");
  __syncthreads();

  for (int c = 0; c < NCHUNK; ++c) {
    if (c + 1 < NCHUNK)                              // prefetch next chunk
      async_stage_chunk(gbase + (uint64_t)(c + 1) * CHUNK_BYTES,
                        ((c + 1) & 1) ? lds1 : lds0, tid);

    const float* s = sx[c & 1];
    const int jbase = c * CHUNK;
    for (int j = 0; j < CHUNK; ++j) {
      float dx = s[j * 3 + 0] - qx;
      float dy = s[j * 3 + 1] - qy;
      float dz = s[j * 3 + 2] - qz;
      float d2 = fmaf(dx, dx, fmaf(dy, dy, dz * dz));
      if (d2 < bd[KNN_K - 1]) {
        float dd = d2; int ii = jbase + j;
#pragma unroll
        for (int st = 0; st < KNN_K; ++st) {
          bool sw = dd < bd[st];
          float td = bd[st]; int ti = bi[st];
          bd[st] = sw ? dd : td;  bi[st] = sw ? ii : ti;
          dd     = sw ? td : dd;  ii     = sw ? ti : ii;
        }
      }
    }

    asm volatile("s_wait_asynccnt 0" ::: "memory");  // next buffer resident
    __syncthreads();                                  // visible to all waves
  }

#pragma unroll
  for (int k = 0; k < KNN_K; ++k) idx_out[(size_t)m * KNN_K + k] = bi[k];
  weight_out[m] = bd[0] > 0.03f ? 10.0f : 1.0f;
}

// ----------------------------------------------------------------------------
// Kernel 3: gather + MLP + weighted reduce. One wave processes 8 queries with
// all 8 W2 B-operands resident (64 VGPRs). h = relu(g_xyz@W1+b1) is built
// per-lane directly in the f16 A-register layout; 8 WMMAs against W2; then
// sum_k (pw+b2)*g_feature via 8 FMAs + one shfl_xor(16) k-half fold.
// ----------------------------------------------------------------------------
#define QPW 8
__global__ void __launch_bounds__(256)
gather_mlp_wmma(const float* __restrict__ f, const float* __restrict__ xyz1,
                const float* __restrict__ xyz2, const int* __restrict__ idx,
                const _Float16* __restrict__ W2P, const float* __restrict__ W1,
                const float* __restrict__ b1, const float* __restrict__ b2,
                float* __restrict__ out) {
  __shared__ float sW1[3 * DIM];
  __shared__ float sb1[DIM];
  __shared__ float sb2[DIM];
  if (threadIdx.x < 192) sW1[threadIdx.x] = W1[threadIdx.x];
  if (threadIdx.x < 64) { sb1[threadIdx.x] = b1[threadIdx.x]; sb2[threadIdx.x] = b2[threadIdx.x]; }
  __syncthreads();

  const int lane = threadIdx.x & 31;
  const int hi = lane >> 4, n = lane & 15;
  const int wave = (blockIdx.x * blockDim.x + threadIdx.x) >> 5;

  v16h bmat[8];
#pragma unroll
  for (int qt = 0; qt < 8; ++qt)
    bmat[qt] = *(const v16h*)(W2P + ((size_t)qt * 32 + lane) * 16);

  for (int qq = 0; qq < QPW; ++qq) {
    const int m = wave * QPW + qq;
    const int b = m >> 13;
    const int nbr = idx[(size_t)m * KNN_K + n];      // A row owned by this lane
    const float qx = xyz2[(size_t)m * 3 + 0];
    const float qy = xyz2[(size_t)m * 3 + 1];
    const float qz = xyz2[(size_t)m * 3 + 2];
    const float px = xyz1[(size_t)(b * NPTS + nbr) * 3 + 0] - qx;
    const float py = xyz1[(size_t)(b * NPTS + nbr) * 3 + 1] - qy;
    const float pz = xyz1[(size_t)(b * NPTS + nbr) * 3 + 2] - qz;

    v16h a[2];
#pragma unroll
    for (int q = 0; q < 2; ++q) {
#pragma unroll
      for (int v = 0; v < 16; ++v) {
        const int j = q * 32 + (v < 8 ? v + hi * 8 : v + 8 + hi * 8);
        float h = fmaf(px, sW1[j], fmaf(py, sW1[64 + j], fmaf(pz, sW1[128 + j], sb1[j])));
        a[q][v] = (_Float16)fmaxf(h, 0.0f);
      }
    }

    int idxr[8];
#pragma unroll
    for (int r = 0; r < 8; ++r) idxr[r] = idx[(size_t)m * KNN_K + r + hi * 8];

#pragma unroll
    for (int t = 0; t < 4; ++t) {
      v8f acc = {};
      acc = WMMA_F16(a[0], bmat[t], acc);
      acc = WMMA_F16(a[1], bmat[4 + t], acc);
      const int c = t * 16 + n;
      const float b2c = sb2[c];
      float s = 0.0f;
#pragma unroll
      for (int r = 0; r < 8; ++r) {
        const float gf = f[(size_t)(b * NPTS + idxr[r]) * DIM + c];
        s = fmaf(acc[r] + b2c, gf, s);
      }
      s += __shfl_xor(s, 16, 32);                    // fold k-rows 8..15 half
      if (!hi) out[(size_t)m * DIM + c] = s * 0.25f; // 1/sqrt(16)
    }
  }
}

// ----------------------------------------------------------------------------
extern "C" void kernel_launch(void* const* d_in, const int* in_sizes, int n_in,
                              void* d_out, int out_size, void* d_ws, size_t ws_size,
                              hipStream_t stream) {
  const float* feature1 = (const float*)d_in[0];
  const float* xyz1     = (const float*)d_in[1];
  const float* xyz2     = (const float*)d_in[2];
  const float* Wp       = (const float*)d_in[3];
  const float* bp       = (const float*)d_in[4];
  const float* W1       = (const float*)d_in[5];
  const float* b1       = (const float*)d_in[6];
  const float* W2       = (const float*)d_in[7];
  const float* b2       = (const float*)d_in[8];

  float* out_feat = (float*)d_out;                         // (B, M, 64)
  float* out_w    = out_feat + (size_t)NB * NPTS * DIM;    // (B, M)

  char* ws = (char*)d_ws;
  float*     fbuf = (float*)(ws + WS_F);
  int*       idx  = (int*)(ws + WS_IDX);
  _Float16*  WpP  = (_Float16*)(ws + WS_WPP);
  _Float16*  W2P  = (_Float16*)(ws + WS_W2P);

  pack_b16<<<16, 256, 0, stream>>>(Wp, WpP);
  pack_b16<<<16, 256, 0, stream>>>(W2, W2P);
  project_wmma<<<128, 256, 0, stream>>>(feature1, WpP, bp, fbuf);       // 1024 waves
  knn_bruteforce<<<64, 256, 0, stream>>>(xyz1, xyz2, idx, out_w);       // 1 lane/query
  gather_mlp_wmma<<<256, 256, 0, stream>>>(fbuf, xyz1, xyz2, idx, W2P,
                                           W1, b1, b2, out_feat);       // 8 queries/wave
  (void)in_sizes; (void)n_in; (void)out_size; (void)ws_size;
}
